// EffAttention_9680856285419
// MI455X (gfx1250) — matrix operand order repaired and checked by reading, not compiled, not run on hardware
//
#include <hip/hip_runtime.h>

// ---------------------------------------------------------------- types ----
typedef __attribute__((ext_vector_type(16))) __bf16 v16bf;
typedef __attribute__((ext_vector_type(8)))  float  v8f;

union AFrag { v16bf v; unsigned u[8]; };

#define C_DIM 768
#define NHEAD 12
#define HD    64
#define S_DIM 4097
#define B_DIM 8
#define MROWS (B_DIM * S_DIM)   // 32776
#define NWIN  16
#define TOK   257               // tokens per window (1 cls + 256)
#define KSTR  72                // LDS row stride (bf16) for K tile: conflict-free, 16B-aligned rows
#define VTSTR 104               // LDS row stride (bf16) for V^T tile: conflict-free, 16B-aligned rows
#define CHUNK 96                // keys staged per LDS chunk

// -------------------------------------------------------------- helpers ----
static __device__ inline __bf16 ea_f2bf(float f) {
  union { float f; unsigned u; } a; a.f = f;
  unsigned r = a.u + 0x7FFFu + ((a.u >> 16) & 1u);   // round-to-nearest-even
  union { unsigned short s; __bf16 b; } o; o.s = (unsigned short)(r >> 16);
  return o.b;
}
static __device__ inline __bf16 ea_us2bf(unsigned short s) {
  union { unsigned short s; __bf16 b; } u; u.s = s; return u.b;
}
// async 16B global -> LDS copy (Tensor/async data path, ASYNCcnt-tracked)
static __device__ inline void ea_async_b128(const void* lds_ptr, const void* gptr) {
  unsigned lds = (unsigned)(size_t)lds_ptr;                 // LDS byte offset (addr[31:0])
  unsigned long long ga = (unsigned long long)gptr;
  asm volatile("global_load_async_to_lds_b128 %0, %1, off" :: "v"(lds), "v"(ga) : "memory");
}
static __device__ inline void ea_wait_async() {
  asm volatile("s_wait_asynccnt 0x0" ::: "memory");
}
// window token t (1..256) -> sequence position in [1, 4097)
static __device__ inline int ea_tok_pos(int w, int t) {
  int wy = w >> 2, wx = w & 3;
  int i = t - 1;
  int r = i >> 4, c = i & 15;
  return 1 + ((wy * 16 + r) << 6) + (wx * 16 + c);
}

// --------------------------------------------------- tiny utility kernels ----
__global__ void ea_cvt_bf16(const float* __restrict__ src, __bf16* __restrict__ dst, int n) {
  int i = blockIdx.x * blockDim.x + threadIdx.x;
  if (i < n) dst[i] = ea_f2bf(src[i]);
}

__global__ void ea_zero_cls(float* __restrict__ ob) {
  int i = blockIdx.x * blockDim.x + threadIdx.x;
  if (i < B_DIM * C_DIM) {
    int b = i / C_DIM, c = i % C_DIM;
    ob[(size_t)b * S_DIM * C_DIM + c] = 0.0f;
  }
}

// ------------------------------------------------- 128x128x32 WMMA GEMM ----
// out[m][n] = sum_k A[m][k] * W[n][k]  (x @ W^T), K = N = 768.
// A: fp32 row-major (converted to bf16 while staging), W: bf16 row-major [n][k].
template <bool OUT_BF16>
__global__ __launch_bounds__(256)
void ea_gemm(const float* __restrict__ A, const __bf16* __restrict__ W,
             const float* __restrict__ bias, void* __restrict__ out, int M) {
  __shared__ __attribute__((aligned(16))) __bf16 As[128 * 40];  // 128 rows x 32 k (+8 pad)
  __shared__ __attribute__((aligned(16))) __bf16 Ws[128 * 40];  // 128 n    x 32 k (+8 pad)

  const int tid  = threadIdx.x;
  const int wid  = tid >> 5;
  const int lane = tid & 31;
  const int ln   = lane & 15;      // N / M-within-half index
  const int sel  = lane >> 4;      // half-wave select
  const int row0 = blockIdx.x * 128;
  const int col0 = blockIdx.y * 128;

  v8f acc[8];
#pragma unroll
  for (int i = 0; i < 8; ++i) acc[i] = (v8f)0.0f;

  for (int k0 = 0; k0 < C_DIM; k0 += 32) {
    // ---- stage W tile via async global->LDS DMA (bf16, no VGPR round-trip) ----
    for (int i = tid; i < 128 * 2; i += 256) {
      int r = i >> 1, seg = i & 1;
      ea_async_b128(&Ws[r * 40 + seg * 16],
                    W + (size_t)(col0 + r) * C_DIM + k0 + seg * 16);
    }
    // ---- stage A tile (fp32 -> bf16 conversion: must go through VGPRs) ----
    for (int i = tid; i < 128 * 8; i += 256) {
      int r = i >> 3, seg = i & 7;
      int gr = row0 + r;
      float4 vv = make_float4(0.f, 0.f, 0.f, 0.f);
      if (gr < M) vv = *(const float4*)(A + (size_t)gr * C_DIM + k0 + seg * 4);
      __bf16* d = &As[r * 40 + seg * 4];
      d[0] = ea_f2bf(vv.x); d[1] = ea_f2bf(vv.y);
      d[2] = ea_f2bf(vv.z); d[3] = ea_f2bf(vv.w);
    }
    ea_wait_async();
    __syncthreads();

    // ---- A fragment for this wave's 16-row strip ----
    AFrag af;
    {
      const __bf16* ap = &As[(wid * 16 + ln) * 40];
      const int kb = sel * 16;
#pragma unroll
      for (int j = 0; j < 4; ++j) af.u[j]     = *(const unsigned*)(ap + kb + 2 * j);
#pragma unroll
      for (int j = 0; j < 4; ++j) af.u[4 + j] = *(const unsigned*)(ap + 8 + kb + 2 * j);
    }
    // ---- 8 N-tiles: B frag + WMMA each ----
#pragma unroll
    for (int nt = 0; nt < 8; ++nt) {
      AFrag bf;
      const __bf16* bp = &Ws[(nt * 16 + ln) * 40];
      const int kb2 = sel * 16;
#pragma unroll
      for (int j = 0; j < 8; ++j) bf.u[j] = *(const unsigned*)(bp + kb2 + 2 * j);
      acc[nt] = __builtin_amdgcn_wmma_f32_16x16x32_bf16(
          false, af.v, false, bf.v, (short)0, acc[nt], false, false);
    }
    __syncthreads();
  }

  // ---- epilogue: C layout VGPR r <-> M = r + 8*sel, lane ln <-> N ----
#pragma unroll
  for (int nt = 0; nt < 8; ++nt) {
#pragma unroll
    for (int r = 0; r < 8; ++r) {
      int grow = row0 + wid * 16 + r + 8 * sel;
      int gcol = col0 + nt * 16 + ln;
      if (grow < M) {
        float v = acc[nt][r];
        if (OUT_BF16) {
          ((__bf16*)out)[(size_t)grow * C_DIM + gcol] = ea_f2bf(v);
        } else {
          ((float*)out)[(size_t)grow * C_DIM + gcol] = v + bias[gcol];
        }
      }
    }
  }
}

// ------------------------------------------- windowed flash attention ----
// One block per (b, window, head). 288 threads = 9 waves; wave owns strips
// {wid, wid+9} of 16 queries each (288 padded tokens). Keys processed in
// 96-key LDS chunks (K async-copied row-major; V transposed at staging so
// P@V B-fragments are contiguous b128 loads), 32 keys per WMMA group,
// online softmax.
__global__ __launch_bounds__(288)
void ea_attn(const __bf16* __restrict__ qb, const __bf16* __restrict__ kbuf,
             const __bf16* __restrict__ vbuf, float* __restrict__ ob) {
  __shared__ __attribute__((aligned(16))) __bf16 Ks[CHUNK * KSTR];   // [key][d]
  __shared__ __attribute__((aligned(16))) __bf16 VsT[HD * VTSTR];    // [d][key]
  __shared__ __attribute__((aligned(16))) __bf16 Ps[9 * 16 * 40];    // per-wave P tile

  const int blk = blockIdx.x;
  const int h = blk % NHEAD;
  const int w = (blk / NHEAD) % NWIN;
  const int b = blk / (NHEAD * NWIN);

  const int tid  = threadIdx.x;
  const int wid  = tid >> 5;
  const int lane = tid & 31;
  const int ln   = lane & 15;
  const int sel  = lane >> 4;

  const int strips[2] = { wid, wid + 9 };

  // ---- load Q fragments straight from global (gathered per window token) ----
  AFrag qf[2][2];
#pragma unroll
  for (int si = 0; si < 2; ++si) {
    const int srow = strips[si] * 16 + ln;
    const bool valid = srow < TOK;
#pragma unroll
    for (int f = 0; f < 2; ++f)
#pragma unroll
      for (int j = 0; j < 8; ++j) qf[si][f].u[j] = 0u;
    if (valid) {
      const int p = ea_tok_pos(w, srow);
      const __bf16* base = qb + (size_t)(b * S_DIM + p) * C_DIM + h * HD;
      const int kb = sel * 16;
#pragma unroll
      for (int f = 0; f < 2; ++f) {
        const __bf16* qp = base + f * 32;
#pragma unroll
        for (int j = 0; j < 4; ++j) qf[si][f].u[j]     = *(const unsigned*)(qp + kb + 2 * j);
#pragma unroll
        for (int j = 0; j < 4; ++j) qf[si][f].u[4 + j] = *(const unsigned*)(qp + 8 + kb + 2 * j);
      }
    }
  }

  // ---- online softmax state (per lane: rows r + 8*sel of each strip) ----
  float m_run[2][8], l_run[2][8];
  v8f   Oacc[2][4];
#pragma unroll
  for (int si = 0; si < 2; ++si) {
#pragma unroll
    for (int r = 0; r < 8; ++r) { m_run[si][r] = -3.0e38f; l_run[si][r] = 0.0f; }
#pragma unroll
    for (int nt = 0; nt < 4; ++nt) Oacc[si][nt] = (v8f)0.0f;
  }

  __bf16* pw = &Ps[wid * 640];  // this wave's 16x32 (stride 40) P scratch

  for (int kc = 0; kc < 3; ++kc) {          // 3 chunks of 96 keys = 288
    __syncthreads();
    // ---- stage K (async DMA, row-major) + V^T (transposed scatter) ----
    for (int i = tid; i < CHUNK * 8; i += 288) {
      int r = i >> 3, seg = i & 7;
      int key = kc * CHUNK + r;
      uint4 vv = make_uint4(0u, 0u, 0u, 0u);
      if (key < TOK) {
        const int p = ea_tok_pos(w, key);
        const size_t off = (size_t)(b * S_DIM + p) * C_DIM + h * HD + seg * 8;
        ea_async_b128(&Ks[r * KSTR + seg * 8], kbuf + off);
        vv = *(const uint4*)(vbuf + off);
      } else {
        *(uint4*)(&Ks[r * KSTR + seg * 8]) = vv;   // zero-pad K rows past 257
      }
      union { uint4 q; unsigned short s[8]; } uu; uu.q = vv;
#pragma unroll
      for (int j = 0; j < 8; ++j)
        VsT[(seg * 8 + j) * VTSTR + r] = ea_us2bf(uu.s[j]);
    }
    ea_wait_async();
    __syncthreads();

    for (int kk = 0; kk < 3; ++kk) {        // 32-key groups within chunk
      const int keybase = kc * CHUNK + kk * 32;
#pragma unroll
      for (int si = 0; si < 2; ++si) {
        // ---- scores: two 16x16 tiles, K-dim = hd (2 WMMA steps each) ----
        v8f sc0 = (v8f)0.0f, sc1 = (v8f)0.0f;
#pragma unroll
        for (int dstep = 0; dstep < 2; ++dstep) {
#pragma unroll
          for (int tile = 0; tile < 2; ++tile) {
            AFrag bk;
            const __bf16* kp = &Ks[(kk * 32 + tile * 16 + ln) * KSTR + dstep * 32 + sel * 16];
#pragma unroll
            for (int j = 0; j < 8; ++j) bk.u[j] = *(const unsigned*)(kp + 2 * j);
            if (tile == 0)
              sc0 = __builtin_amdgcn_wmma_f32_16x16x32_bf16(
                  false, qf[si][dstep].v, false, bk.v, (short)0, sc0, false, false);
            else
              sc1 = __builtin_amdgcn_wmma_f32_16x16x32_bf16(
                  false, qf[si][dstep].v, false, bk.v, (short)0, sc1, false, false);
          }
        }
        // ---- scale + key mask ----
        const float scale = 0.125f;  // 1/sqrt(64)
        const int key0 = keybase + ln, key1 = keybase + 16 + ln;
#pragma unroll
        for (int r = 0; r < 8; ++r) {
          sc0[r] = (key0 < TOK) ? sc0[r] * scale : -1.0e30f;
          sc1[r] = (key1 < TOK) ? sc1[r] * scale : -1.0e30f;
        }
        // ---- row max over 32 keys (reduce across 16 lanes of this half) ----
        float rm[8], fr[8], rs[8];
#pragma unroll
        for (int r = 0; r < 8; ++r) rm[r] = fmaxf(sc0[r], sc1[r]);
#pragma unroll
        for (int m = 1; m < 16; m <<= 1)
#pragma unroll
          for (int r = 0; r < 8; ++r) rm[r] = fmaxf(rm[r], __shfl_xor(rm[r], m, 16));
#pragma unroll
        for (int r = 0; r < 8; ++r) {
          float mn = fmaxf(m_run[si][r], rm[r]);
          fr[r] = __expf(m_run[si][r] - mn);
          m_run[si][r] = mn;
        }
        // ---- exponentiate + row sum ----
#pragma unroll
        for (int r = 0; r < 8; ++r) {
          float p0 = __expf(sc0[r] - m_run[si][r]);
          float p1 = __expf(sc1[r] - m_run[si][r]);
          sc0[r] = p0; sc1[r] = p1; rs[r] = p0 + p1;
        }
#pragma unroll
        for (int m = 1; m < 16; m <<= 1)
#pragma unroll
          for (int r = 0; r < 8; ++r) rs[r] += __shfl_xor(rs[r], m, 16);
#pragma unroll
        for (int r = 0; r < 8; ++r) l_run[si][r] = l_run[si][r] * fr[r] + rs[r];
        // ---- rescale O accumulators ----
#pragma unroll
        for (int nt = 0; nt < 4; ++nt)
#pragma unroll
          for (int r = 0; r < 8; ++r) Oacc[si][nt][r] *= fr[r];
        // ---- C-layout P -> LDS (row-major 16x32) -> A-frag ----
#pragma unroll
        for (int r = 0; r < 8; ++r) {
          int mrow = r + 8 * sel;
          pw[mrow * 40 + ln]      = ea_f2bf(sc0[r]);
          pw[mrow * 40 + 16 + ln] = ea_f2bf(sc1[r]);
        }
        AFrag pf;
        {
          const __bf16* pa = pw + ln * 40;
          const int kb = sel * 16;
#pragma unroll
          for (int j = 0; j < 4; ++j) pf.u[j]     = *(const unsigned*)(pa + kb + 2 * j);
#pragma unroll
          for (int j = 0; j < 4; ++j) pf.u[4 + j] = *(const unsigned*)(pa + 8 + kb + 2 * j);
        }
        // ---- O += P @ V: V^T rows give 16 consecutive keys -> 2x b128 ----
#pragma unroll
        for (int nt = 0; nt < 4; ++nt) {
          AFrag bv;
          const __bf16* vp = &VsT[(nt * 16 + ln) * VTSTR + kk * 32 + sel * 16];
          uint4 a0 = *(const uint4*)vp;
          uint4 a1 = *(const uint4*)(vp + 8);
          bv.u[0] = a0.x; bv.u[1] = a0.y; bv.u[2] = a0.z; bv.u[3] = a0.w;
          bv.u[4] = a1.x; bv.u[5] = a1.y; bv.u[6] = a1.z; bv.u[7] = a1.w;
          Oacc[si][nt] = __builtin_amdgcn_wmma_f32_16x16x32_bf16(
              false, pf.v, false, bv.v, (short)0, Oacc[si][nt], false, false);
        }
      }
    }
  }

  // ---- epilogue: normalize and scatter to o buffer (f32, flat layout) ----
#pragma unroll
  for (int si = 0; si < 2; ++si) {
#pragma unroll
    for (int r = 0; r < 8; ++r) {
      const int t = strips[si] * 16 + r + 8 * sel;
      if (t >= TOK) continue;
      const float inv = 1.0f / l_run[si][r];
#pragma unroll
      for (int nt = 0; nt < 4; ++nt) {
        const int col = h * HD + nt * 16 + ln;
        const float val = Oacc[si][nt][r] * inv;
        if (t == 0) {
          atomicAdd(&ob[(size_t)b * S_DIM * C_DIM + col], val * (1.0f / 16.0f));
        } else {
          const int p = ea_tok_pos(w, t);
          ob[(size_t)(b * S_DIM + p) * C_DIM + col] = val;
        }
      }
    }
  }
}

// ---------------------------------------------------------------- launch ----
extern "C" void kernel_launch(void* const* d_in, const int* in_sizes, int n_in,
                              void* d_out, int out_size, void* d_ws, size_t ws_size,
                              hipStream_t stream) {
  const float* x  = (const float*)d_in[0];
  const float* wq = (const float*)d_in[1];
  const float* wk = (const float*)d_in[2];
  const float* wv = (const float*)d_in[3];
  const float* wp = (const float*)d_in[4];
  const float* bp = (const float*)d_in[5];

  const size_t WN  = (size_t)C_DIM * C_DIM;       // 589824
  const size_t MRC = (size_t)MROWS * C_DIM;       // 25,171,968

  __bf16* wqb = (__bf16*)d_ws;
  __bf16* wkb = wqb + WN;
  __bf16* wvb = wkb + WN;
  __bf16* wpb = wvb + WN;
  __bf16* qb  = wpb + WN;
  __bf16* kb  = qb + MRC;
  __bf16* vb  = kb + MRC;
  float*  ob  = (float*)(vb + MRC);

  const int cvtb = (int)((WN + 255) / 256);
  ea_cvt_bf16<<<cvtb, 256, 0, stream>>>(wq, wqb, (int)WN);
  ea_cvt_bf16<<<cvtb, 256, 0, stream>>>(wk, wkb, (int)WN);
  ea_cvt_bf16<<<cvtb, 256, 0, stream>>>(wv, wvb, (int)WN);
  ea_cvt_bf16<<<cvtb, 256, 0, stream>>>(wp, wpb, (int)WN);

  ea_zero_cls<<<(B_DIM * C_DIM + 255) / 256, 256, 0, stream>>>(ob);

  dim3 gg((MROWS + 127) / 128, C_DIM / 128);
  ea_gemm<true><<<gg, 256, 0, stream>>>(x, wqb, nullptr, (void*)qb, MROWS);
  ea_gemm<true><<<gg, 256, 0, stream>>>(x, wkb, nullptr, (void*)kb, MROWS);
  ea_gemm<true><<<gg, 256, 0, stream>>>(x, wvb, nullptr, (void*)vb, MROWS);

  ea_attn<<<B_DIM * NWIN * NHEAD, 288, 0, stream>>>(qb, kb, vb, ob);

  ea_gemm<false><<<gg, 256, 0, stream>>>(ob, wpb, bp, d_out, MROWS);
}
